// LlamaAttention_55722905698867
// MI455X (gfx1250) — compile-verified, matrix operand order checked by reading, not run on hardware
//
#include <hip/hip_runtime.h>
#include <hip/hip_bf16.h>
#include <math.h>
#include <stdint.h>

// ---- problem constants (Llama-2-7B-like GQA prefill) ----
#define H_    4096
#define NH_   32
#define NKV_  8
#define HD_   128
#define B_    4
#define S_    1024
#define T_    (B_ * S_)
#define REP_  (NH_ / NKV_)

typedef __bf16 bf16;
typedef __attribute__((ext_vector_type(16))) __bf16 bf16x16;
typedef __attribute__((ext_vector_type(8)))  __bf16 bf16x8;
typedef __attribute__((ext_vector_type(8)))  float  f32x8;

union FragB { bf16x16 v; bf16x8 h[2]; };

// Load a 16x32 bf16 fragment (A or B operand of v_wmma_f32_16x16x32_bf16).
// Per CDNA5 ISA 7.12.2: lanes 0-15 hold row l with K {0..7,16..23},
// lanes 16-31 hold row l with K {8..15,24..31}. Two b128 loads per lane.
__device__ __forceinline__ bf16x16 load_frag_nt(const bf16* base, int ld, int lane) {
  const int l = lane & 15, g = lane >> 4;
  const bf16* p = base + (size_t)l * ld + g * 8;
  FragB f;
  f.h[0] = *(const bf16x8*)(p);
  f.h[1] = *(const bf16x8*)(p + 16);
  return f.v;
}

__device__ __forceinline__ f32x8 wmma_bf16(bf16x16 a, bf16x16 b, f32x8 c) {
  return __builtin_amdgcn_wmma_f32_16x16x32_bf16(false, a, false, b,
                                                 (short)0, c, false, false);
}

// Async copy 16 bytes global -> LDS (ASYNCcnt-tracked, no VGPR staging).
__device__ __forceinline__ void async_b128(uint32_t lds_byte_addr, const void* gptr) {
  asm volatile("global_load_async_to_lds_b128 %0, %1, off"
               :: "v"(lds_byte_addr), "v"(gptr) : "memory");
}
__device__ __forceinline__ void wait_async0() {
  asm volatile("s_wait_asynccnt 0" ::: "memory");
}

// ---------------- fp32 -> bf16 conversion ----------------
__global__ void cvt_f32_bf16_kernel(const float* __restrict__ in,
                                    bf16* __restrict__ out, size_t n) {
  size_t i = (size_t)blockIdx.x * blockDim.x + threadIdx.x;
  size_t stride = (size_t)gridDim.x * blockDim.x;
  for (; i < n; i += stride) out[i] = (bf16)in[i];
}

// ---------------- C[M,N] = A[M,K] * B[N,K]^T (bf16 in, f32 acc) -------------
// 128x128 block tile, 4 waves (2x2) of 64x64. Operands double-buffered in LDS
// via global_load_async_to_lds_b128, overlapped with WMMA on the current slab.
__global__ __launch_bounds__(128) void gemm_bf16_nt_kernel(
    const bf16* __restrict__ A, const bf16* __restrict__ Bm,
    float* __restrict__ Cf, bf16* __restrict__ Cb, int M, int N, int K) {
  __shared__ __align__(16) bf16 smem[2][2][128 * 32];  // [buf][A|B][row*32+k]
  const int tid  = threadIdx.x;
  const int lane = tid & 31;
  const int wave = tid >> 5;
  const int row0 = blockIdx.y * 128 + (wave >> 1) * 64;
  const int col0 = blockIdx.x * 128 + (wave & 1) * 64;

  // Each of the 128 threads stages one 64B A-row and one 64B B-row per slab.
  const bf16* garow = A  + (size_t)(blockIdx.y * 128 + tid) * K;
  const bf16* gbrow = Bm + (size_t)(blockIdx.x * 128 + tid) * K;

  auto stage = [&](int k0, int buf) {
    const uint32_t la = (uint32_t)(uintptr_t)&smem[buf][0][tid * 32];
    const uint32_t lb = (uint32_t)(uintptr_t)&smem[buf][1][tid * 32];
#pragma unroll
    for (int c = 0; c < 4; ++c) {
      async_b128(la + c * 16, garow + k0 + c * 8);
      async_b128(lb + c * 16, gbrow + k0 + c * 8);
    }
  };

  f32x8 acc[4][4] = {};
  const int nslab = K / 32;
  stage(0, 0);
  wait_async0();
  __syncthreads();

  for (int s = 0; s < nslab; ++s) {
    const int buf = s & 1;
    if (s + 1 < nslab) stage((s + 1) * 32, buf ^ 1);  // overlap copy w/ math

    const bf16* As = &smem[buf][0][(wave >> 1) * 64 * 32];
    const bf16* Bs = &smem[buf][1][(wave &  1) * 64 * 32];
    bf16x16 af[4], bfr[4];
#pragma unroll
    for (int i = 0; i < 4; ++i) af[i]  = load_frag_nt(As + 16 * i * 32, 32, lane);
#pragma unroll
    for (int j = 0; j < 4; ++j) bfr[j] = load_frag_nt(Bs + 16 * j * 32, 32, lane);
#pragma unroll
    for (int i = 0; i < 4; ++i)
#pragma unroll
      for (int j = 0; j < 4; ++j)
        acc[i][j] = wmma_bf16(af[i], bfr[j], acc[i][j]);

    wait_async0();     // next slab landed in LDS
    __syncthreads();   // drains DScnt, so WAR on buf^1 next iter is safe
  }

  // D layout: VGPR r, lanes 0-15 -> (M=r, N=lane); lanes 16-31 -> (M=r+8, N=lane-16)
  const int l = lane & 15, g = lane >> 4;
#pragma unroll
  for (int i = 0; i < 4; ++i)
#pragma unroll
    for (int j = 0; j < 4; ++j)
#pragma unroll
      for (int r = 0; r < 8; ++r) {
        const int row = row0 + 16 * i + r + 8 * g;
        const int col = col0 + 16 * j + l;
        if (Cf) Cf[(size_t)row * N + col] = acc[i][j][r];
        else    Cb[(size_t)row * N + col] = (bf16)acc[i][j][r];
      }
}

// ---------------- RoPE (in place on bf16 q/k), faithful to reference -------
__global__ void rope_kernel(bf16* __restrict__ qk,
                            const int* __restrict__ positions, int nheads) {
  size_t idx = (size_t)blockIdx.x * blockDim.x + threadIdx.x;
  const size_t total = (size_t)T_ * nheads * (HD_ / 2);
  if (idx >= total) return;
  const int d  = (int)(idx % (HD_ / 2));
  size_t th    = idx / (HD_ / 2);
  const int hh = (int)(th % nheads);
  const int t  = (int)(th / nheads);
  int pos = positions[t];
  pos = pos < 0 ? 0 : (pos > S_ - 1 ? S_ - 1 : pos);
  const float p = (float)pos;
  bf16* base = qk + ((size_t)t * nheads + hh) * HD_;
  const float x0 = (float)base[d];
  const float x1 = (float)base[d + 64];
  const float a0 = p * __powf(10000.0f, -(float)(d >> 1) / 64.0f);
  const float a1 = p * __powf(10000.0f, -(float)(32 + (d >> 1)) / 64.0f);
  base[d]      = (bf16)(x0 * __cosf(a0) - x1 * __sinf(a0));
  base[d + 64] = (bf16)(x1 * __cosf(a1) + x0 * __sinf(a1));
}

// ---------------- V relayout: [T, NKV*HD] -> [B,NKV,HD,S] (K-contig for PV) -
__global__ void vtrans_kernel(const bf16* __restrict__ v, bf16* __restrict__ vt) {
  size_t idx = (size_t)blockIdx.x * blockDim.x + threadIdx.x;
  if (idx >= (size_t)T_ * NKV_ * HD_) return;
  const int c = (int)(idx % (NKV_ * HD_));
  const size_t t = idx / (NKV_ * HD_);
  const int b = (int)(t / S_), s = (int)(t % S_);
  const int kvh = c >> 7, d = c & 127;
  vt[(((size_t)b * NKV_ + kvh) * HD_ + d) * S_ + s] = v[idx];
}

// ---------------- flash attention (causal, GQA) -----------------------------
__global__ __launch_bounds__(128) void attn_kernel(
    const bf16* __restrict__ Q, const bf16* __restrict__ Kb,
    const bf16* __restrict__ Vt, bf16* __restrict__ Ab) {
  __shared__ __align__(16) bf16 pstage[4][16 * 48];  // per-wave P staging
  const int lane = threadIdx.x & 31;
  const int wave = threadIdx.x >> 5;
  const int b = blockIdx.z, h = blockIdx.y;
  const int kvh = h / REP_;
  const int q0 = blockIdx.x * 64 + wave * 16;
  const int l = lane & 15, g = lane >> 4;
  const float scale = 0.08838834764831845f;  // 1/sqrt(128)

  bf16x16 qf[4];
#pragma unroll
  for (int i = 0; i < 4; ++i)
    qf[i] = load_frag_nt(Q + (size_t)(b * S_ + q0) * H_ + h * HD_ + i * 32, H_, lane);

  f32x8 o[8] = {};
  float mrow[8], lrow[8];
#pragma unroll
  for (int r = 0; r < 8; ++r) { mrow[r] = -__builtin_inff(); lrow[r] = 0.0f; }

  bf16* pls = &pstage[wave][0];
  const int kmax = q0 + 16;
  for (int kt = 0; kt < kmax; kt += 32) {
    if (kt + 32 < kmax)  // global_prefetch of next K tile
      __builtin_prefetch(Kb + (size_t)(b * S_ + kt + 32 + l) * (NKV_ * HD_) + kvh * HD_, 0, 0);
    f32x8 s0 = {}, s1 = {};
#pragma unroll
    for (int i = 0; i < 4; ++i) {
      bf16x16 kf = load_frag_nt(
          Kb + (size_t)(b * S_ + kt) * (NKV_ * HD_) + kvh * HD_ + i * 32,
          NKV_ * HD_, lane);
      s0 = wmma_bf16(qf[i], kf, s0);
    }
    if (kt + 16 < kmax) {
#pragma unroll
      for (int i = 0; i < 4; ++i) {
        bf16x16 kf = load_frag_nt(
            Kb + (size_t)(b * S_ + kt + 16) * (NKV_ * HD_) + kvh * HD_ + i * 32,
            NKV_ * HD_, lane);
        s1 = wmma_bf16(qf[i], kf, s1);
      }
    }
    float p0[8], p1[8], corr[8];
#pragma unroll
    for (int r = 0; r < 8; ++r) {
      const int row = q0 + r + 8 * g;
      float v0 = (kt + l <= row)      ? s0[r] * scale : -__builtin_inff();
      float v1 = (kt + 16 + l <= row) ? s1[r] * scale : -__builtin_inff();
      float mx = fmaxf(v0, v1);
#pragma unroll
      for (int off = 1; off < 16; off <<= 1) mx = fmaxf(mx, __shfl_xor(mx, off, 32));
      const float mnew = fmaxf(mrow[r], mx);
      corr[r] = __expf(mrow[r] - mnew);
      p0[r] = __expf(v0 - mnew);
      p1[r] = __expf(v1 - mnew);
      float ps = p0[r] + p1[r];
#pragma unroll
      for (int off = 1; off < 16; off <<= 1) ps += __shfl_xor(ps, off, 32);
      lrow[r] = lrow[r] * corr[r] + ps;
      mrow[r] = mnew;
    }
#pragma unroll
    for (int t = 0; t < 8; ++t)
#pragma unroll
      for (int r = 0; r < 8; ++r) o[t][r] *= corr[r];

    // reshape P (f32 D-layout) -> bf16 A-fragment through LDS
#pragma unroll
    for (int r = 0; r < 8; ++r) {
      pls[(r + 8 * g) * 48 + l]      = (bf16)p0[r];
      pls[(r + 8 * g) * 48 + 16 + l] = (bf16)p1[r];
    }
    asm volatile("s_wait_dscnt 0" ::: "memory");
    FragB pf;
    { const bf16* pp = pls + l * 48 + g * 8;
      pf.h[0] = *(const bf16x8*)pp; pf.h[1] = *(const bf16x8*)(pp + 16); }

    const bf16* vbase = Vt + ((size_t)b * NKV_ + kvh) * HD_ * (size_t)S_;
#pragma unroll
    for (int t = 0; t < 8; ++t) {
      FragB vf;
      const bf16* vp = vbase + (size_t)(t * 16 + l) * S_ + kt + g * 8;
      vf.h[0] = *(const bf16x8*)vp; vf.h[1] = *(const bf16x8*)(vp + 16);
      o[t] = wmma_bf16(pf.v, vf.v, o[t]);
    }
  }

#pragma unroll
  for (int r = 0; r < 8; ++r) {
    const float inv = 1.0f / lrow[r];
    const int row = b * S_ + q0 + r + 8 * g;
#pragma unroll
    for (int t = 0; t < 8; ++t)
      Ab[(size_t)row * H_ + h * HD_ + t * 16 + l] = (bf16)(o[t][r] * inv);
  }
}

// ---------------------------------------------------------------------------
extern "C" void kernel_launch(void* const* d_in, const int* in_sizes, int n_in,
                              void* d_out, int out_size, void* d_ws, size_t ws_size,
                              hipStream_t stream) {
  (void)in_sizes; (void)n_in; (void)out_size; (void)ws_size;
  const float* x        = (const float*)d_in[0];
  const int*  positions = (const int*)d_in[1];
  const float* wq = (const float*)d_in[3];
  const float* wk = (const float*)d_in[4];
  const float* wv = (const float*)d_in[5];
  const float* wo = (const float*)d_in[6];

  char* ws = (char*)d_ws;
  size_t off = 0;
  auto alloc = [&](size_t bytes) -> void* {
    void* p = ws + off;
    off += (bytes + 255) & ~(size_t)255;
    return p;
  };
  bf16* xb   = (bf16*)alloc((size_t)T_ * H_ * 2);
  bf16* wqb  = (bf16*)alloc((size_t)NH_ * HD_ * H_ * 2);
  bf16* wkb  = (bf16*)alloc((size_t)NKV_ * HD_ * H_ * 2);
  bf16* wvb  = (bf16*)alloc((size_t)NKV_ * HD_ * H_ * 2);
  bf16* wob  = (bf16*)alloc((size_t)H_ * NH_ * HD_ * 2);
  bf16* qb   = (bf16*)alloc((size_t)T_ * NH_ * HD_ * 2);
  bf16* kbuf = (bf16*)alloc((size_t)T_ * NKV_ * HD_ * 2);
  bf16* vbuf = (bf16*)alloc((size_t)T_ * NKV_ * HD_ * 2);
  bf16* vt   = (bf16*)alloc((size_t)T_ * NKV_ * HD_ * 2 + 4096);
  bf16* ab   = (bf16*)alloc((size_t)T_ * NH_ * HD_ * 2);

  const int CB = 256, CG = 8192;
  cvt_f32_bf16_kernel<<<CG, CB, 0, stream>>>(x,  xb,  (size_t)T_ * H_);
  cvt_f32_bf16_kernel<<<CG, CB, 0, stream>>>(wq, wqb, (size_t)NH_ * HD_ * H_);
  cvt_f32_bf16_kernel<<<CG, CB, 0, stream>>>(wk, wkb, (size_t)NKV_ * HD_ * H_);
  cvt_f32_bf16_kernel<<<CG, CB, 0, stream>>>(wv, wvb, (size_t)NKV_ * HD_ * H_);
  cvt_f32_bf16_kernel<<<CG, CB, 0, stream>>>(wo, wob, (size_t)H_ * NH_ * HD_);

  gemm_bf16_nt_kernel<<<dim3((NH_ * HD_) / 128, T_ / 128), 128, 0, stream>>>(
      xb, wqb, nullptr, qb, T_, NH_ * HD_, H_);
  gemm_bf16_nt_kernel<<<dim3((NKV_ * HD_) / 128, T_ / 128), 128, 0, stream>>>(
      xb, wkb, nullptr, kbuf, T_, NKV_ * HD_, H_);
  gemm_bf16_nt_kernel<<<dim3((NKV_ * HD_) / 128, T_ / 128), 128, 0, stream>>>(
      xb, wvb, nullptr, vbuf, T_, NKV_ * HD_, H_);

  rope_kernel<<<((size_t)T_ * NH_ * 64 + CB - 1) / CB, CB, 0, stream>>>(qb, positions, NH_);
  rope_kernel<<<((size_t)T_ * NKV_ * 64 + CB - 1) / CB, CB, 0, stream>>>(kbuf, positions, NKV_);

  vtrans_kernel<<<((size_t)T_ * NKV_ * HD_ + CB - 1) / CB, CB, 0, stream>>>(vbuf, vt);

  attn_kernel<<<dim3(S_ / 64, NH_, B_), 128, 0, stream>>>(qb, kbuf, vt, ab);

  gemm_bf16_nt_kernel<<<dim3(H_ / 128, T_ / 128), 128, 0, stream>>>(
      ab, wob, (float*)d_out, nullptr, T_, H_, NH_ * HD_);
}
